// MSHGAT_74251394613678
// MI455X (gfx1250) — compile-verified, matrix operand order checked by reading, not run on hardware
//
#include <hip/hip_runtime.h>
#include <hip/hip_bf16.h>

// Sizes from the reference
#define NN   50000
#define DD   64
#define EE   600000
#define NNZT 800000
#define BT   256

typedef __attribute__((ext_vector_type(16))) __bf16 v16bf;
typedef __attribute__((ext_vector_type(8)))  float  v8f;

// ---------------------------------------------------------------------------
// 16-bit WMMA fragment layout (ISA 7.12.2, 16x32 MxK, wave32):
//   lane = grp*16 + i16 ; fragment element e (0..15):
//     e in 0..7  -> k = 8*grp + e        (one contiguous 8-float segment)
//     e in 8..15 -> k = 16 + 8*grp + e-8 (second contiguous 8-float segment)
// So each lane's A (or row-major B^T) fragment is two 32B segments -> 4x float4.
// ---------------------------------------------------------------------------
__device__ __forceinline__ void cvt8(v16bf& f, int base, float4 a, float4 b) {
    f[base + 0] = (__bf16)a.x; f[base + 1] = (__bf16)a.y;
    f[base + 2] = (__bf16)a.z; f[base + 3] = (__bf16)a.w;
    f[base + 4] = (__bf16)b.x; f[base + 5] = (__bf16)b.y;
    f[base + 6] = (__bf16)b.z; f[base + 7] = (__bf16)b.w;
}

// Y[nrows,M] = X[nrows,K] @ (WT ? W[M,K]^T : W[K,M]) (+ bias[col])
// One wave computes one 16x16 f32 tile via v_wmma_f32_16x16x32_bf16.
template <int K, int M, bool WT>
__global__ void gemm_bf16_wmma(const float* __restrict__ X,
                               const float* __restrict__ W,
                               const float* __restrict__ bias,
                               float* __restrict__ Y, int nrows) {
    const int lane  = threadIdx.x & 31;
    const int wave  = (blockIdx.x * blockDim.x + threadIdx.x) >> 5;
    const int mt    = M / 16;
    const int tileM = wave % mt;
    const int tileN = wave / mt;
    if (tileN * 16 >= nrows) return;          // wave-uniform: EXEC all-ones for WMMA
    const int grp = lane >> 4;                 // half-wave group
    const int i16 = lane & 15;
    const int col = tileM * 16 + i16;

    v8f acc = {};
    const float* xrow = X + (size_t)(tileN * 16 + i16) * K;
    const float* wrow = WT ? (W + (size_t)col * K) : nullptr;
    for (int kk = 0; kk < K; kk += 32) {
        v16bf a, b;
        {   // A: two contiguous 32B segments per lane -> 4 x global_load_b128
            const float4* s0 = (const float4*)(xrow + kk + 8 * grp);
            const float4* s1 = (const float4*)(xrow + kk + 16 + 8 * grp);
            cvt8(a, 0, s0[0], s0[1]);
            cvt8(a, 8, s1[0], s1[1]);
        }
        if (WT) {   // B^T rows are contiguous: same vectorized pattern
            const float4* s0 = (const float4*)(wrow + kk + 8 * grp);
            const float4* s1 = (const float4*)(wrow + kk + 16 + 8 * grp);
            cvt8(b, 0, s0[0], s0[1]);
            cvt8(b, 8, s1[0], s1[1]);
        } else {    // column access, strided by M: scalar loads (W is tiny, L0-hot)
#pragma unroll
            for (int e = 0; e < 8; ++e) {
                b[e]     = (__bf16)W[(size_t)(kk + 8 * grp + e) * M + col];
                b[8 + e] = (__bf16)W[(size_t)(kk + 16 + 8 * grp + e) * M + col];
            }
        }
        acc = __builtin_amdgcn_wmma_f32_16x16x32_bf16(
            /*neg_a=*/false, a, /*neg_b=*/false, b,
            /*c_mod=*/(short)0, acc, /*reuse_a=*/false, /*reuse_b=*/false);
    }
    const float bb = bias ? bias[col] : 0.0f;
#pragma unroll
    for (int r = 0; r < 8; ++r) {             // D layout: row = 8*grp + r, col = i16
        const int row = tileN * 16 + 8 * grp + r;
        Y[(size_t)row * M + col] = acc[r] + bb;
    }
}

// ---------------------------------------------------------------------------
// Degree / normalization (deg includes self loop; dinv = rsqrt(deg) in place)
// ---------------------------------------------------------------------------
__global__ void deg_init(float* deg) {
    int i = blockIdx.x * blockDim.x + threadIdx.x;
    if (i < NN) deg[i] = 1.0f;                // self loop
}
__global__ void deg_count(const int* __restrict__ dst, float* deg) {
    int e = blockIdx.x * blockDim.x + threadIdx.x;
    if (e < EE) atomicAdd(&deg[dst[e]], 1.0f);
}
__global__ void deg_finalize(float* deg) {
    int i = blockIdx.x * blockDim.x + threadIdx.x;
    if (i < NN) deg[i] = rsqrtf(deg[i]);
}

// out[n,m] = bias[m] + dinv[n]^2 * xw[n,m]   (self-loop term + bias), float4
template <int M>
__global__ void gcn_init(const float* __restrict__ xw, const float* __restrict__ bias,
                         const float* __restrict__ dinv, float* __restrict__ out) {
    long i4 = (long)blockIdx.x * blockDim.x + threadIdx.x;
    if (i4 >= (long)NN * (M / 4)) return;
    int n  = (int)(i4 >> (M == 128 ? 5 : 4));
    int m4 = (int)(i4 & (M / 4 - 1));
    float di = dinv[n];
    float d2 = di * di;
    float4 v = ((const float4*)xw)[i4];
    const float4 bb = ((const float4*)bias)[m4];
    float4 o = { bb.x + d2 * v.x, bb.y + d2 * v.y, bb.z + d2 * v.z, bb.w + d2 * v.w };
    ((float4*)out)[i4] = o;
}

// out[dst] += dinv[src]*dinv[dst]*xw[src]; LPE lanes per edge, float4 per lane
template <int LPE>
__global__ void gcn_scatter(const float* __restrict__ xw, const float* __restrict__ dinv,
                            const int* __restrict__ src, const int* __restrict__ dst,
                            float* __restrict__ out) {
    long tid = (long)blockIdx.x * blockDim.x + threadIdx.x;
    long e = tid / LPE;
    int  l = (int)(tid % LPE);
    if (e >= EE) return;
    const int M = LPE * 4;
    int s = src[e], t = dst[e];
    float nrm = dinv[s] * dinv[t];
    float4 v = ((const float4*)(xw + (size_t)s * M))[l];
    float* o = out + (size_t)t * M + 4 * l;
    atomicAdd(o + 0, nrm * v.x);
    atomicAdd(o + 1, nrm * v.y);
    atomicAdd(o + 2, nrm * v.z);
    atomicAdd(o + 3, nrm * v.w);
}

// ---------------------------------------------------------------------------
// BatchNorm, coalesced two-phase reduction:
//  phase 1: 16 lanes cover a full 256B row as float4; LDS-reduce 16 row groups;
//           atomicAdd 64 sums + 64 sumsq into global accum
//  phase 2: finalize mu / rstd
// ---------------------------------------------------------------------------
__global__ void bn_partial(const float* __restrict__ h2, float* __restrict__ accum) {
    __shared__ float lsum[BT * 4];
    __shared__ float lsq[BT * 4];
    const int t    = threadIdx.x;
    const int colg = t & 15;   // 16 col groups x 4 floats = 64 columns
    const int rowg = t >> 4;   // 16 row groups
    float4 s = {0, 0, 0, 0}, q = {0, 0, 0, 0};
    for (int r = blockIdx.x * 16 + rowg; r < NN; r += gridDim.x * 16) {
        float4 v = *(const float4*)(h2 + (size_t)r * DD + colg * 4);
        s.x += v.x; s.y += v.y; s.z += v.z; s.w += v.w;
        q.x += v.x * v.x; q.y += v.y * v.y; q.z += v.z * v.z; q.w += v.w * v.w;
    }
    ((float4*)lsum)[t] = s;
    ((float4*)lsq)[t]  = q;
    __syncthreads();
    if (t < DD) {              // one thread per column
        int cg = t >> 2, j = t & 3;
        float ts = 0.f, tq = 0.f;
#pragma unroll
        for (int rg = 0; rg < 16; ++rg) {
            ts += lsum[(rg * 16 + cg) * 4 + j];
            tq += lsq[(rg * 16 + cg) * 4 + j];
        }
        atomicAdd(&accum[t], ts);
        atomicAdd(&accum[DD + t], tq);
    }
}

__global__ void bn_finalize(const float* __restrict__ accum, float* __restrict__ stats) {
    int c = threadIdx.x;
    if (c < DD) {
        float mu  = accum[c] / (float)NN;
        float var = accum[DD + c] / (float)NN - mu * mu;
        stats[c]      = mu;
        stats[DD + c] = rsqrtf(var + 1e-5f);
    }
}

__global__ void bn_apply(const float* __restrict__ h2, const float* __restrict__ stats,
                         const float* __restrict__ gamma, const float* __restrict__ beta,
                         const float* __restrict__ hbias,
                         float* __restrict__ hidden, float* __restrict__ xrelu) {
    long i = (long)blockIdx.x * blockDim.x + threadIdx.x;
    if (i >= (long)NN * DD) return;
    int m = (int)(i & (DD - 1));
    float h = (h2[i] - stats[m]) * stats[DD + m] * gamma[m] + beta[m];
    hidden[i] = h;
    xrelu[i]  = fmaxf(h, 0.0f) + hbias[m];
}

// ---------------------------------------------------------------------------
// Hypergraph sparse pass: out[sidx[e]] += val[e] * xin[gidx[e]]  (D=64)
// 16 lanes per nnz, float4 per lane.
// ---------------------------------------------------------------------------
__global__ void hg_scatter(const float* __restrict__ xin, const float* __restrict__ val,
                           const int* __restrict__ gidx, const int* __restrict__ sidx,
                           float* __restrict__ out) {
    long tid = (long)blockIdx.x * blockDim.x + threadIdx.x;
    long e = tid >> 4;
    int  l = (int)(tid & 15);
    if (e >= NNZT) return;
    int g = gidx[e], s = sidx[e];
    float v = val[e];
    float4 d = ((const float4*)(xin + (size_t)g * DD))[l];
    float* o = out + (size_t)s * DD + 4 * l;
    atomicAdd(o + 0, v * d.x);
    atomicAdd(o + 1, v * d.y);
    atomicAdd(o + 2, v * d.z);
    atomicAdd(o + 3, v * d.w);
}

__global__ void vfill(float* __restrict__ p, float v, long n) {
    long i = (long)blockIdx.x * blockDim.x + threadIdx.x;
    if (i < n) p[i] = v;
}

__global__ void softmax_rows(float* __restrict__ x) {
    int n = blockIdx.x * blockDim.x + threadIdx.x;
    if (n >= NN) return;
    float* row = x + (size_t)n * DD;
    float mx = -3.4e38f;
    for (int k = 0; k < DD; ++k) mx = fmaxf(mx, row[k]);
    float s = 0.f;
    for (int k = 0; k < DD; ++k) { float e = __expf(row[k] - mx); row[k] = e; s += e; }
    float inv = 1.0f / s;
    for (int k = 0; k < DD; ++k) row[k] *= inv;
}

// Fusion gate: s_i = tanh(t_i) . w2 + b2 ; softmax over i in {0,1}
__global__ void fusion(const float* __restrict__ hidden, const float* __restrict__ subn,
                       const float* __restrict__ t1, const float* __restrict__ t2,
                       const float* __restrict__ w2, const float* __restrict__ b2,
                       float* __restrict__ out) {
    int n = blockIdx.x * blockDim.x + threadIdx.x;
    if (n >= NN) return;
    size_t base = (size_t)n * DD;
    float s0 = b2[0], s1 = b2[0];
    for (int k = 0; k < DD; ++k) {
        float w = w2[k];
        s0 += tanhf(t1[base + k]) * w;
        s1 += tanhf(t2[base + k]) * w;
    }
    float m  = fmaxf(s0, s1);
    float e0 = __expf(s0 - m), e1 = __expf(s1 - m);
    float inv = 1.0f / (e0 + e1);
    float w0 = e0 * inv, w1 = e1 * inv;
    for (int k = 0; k < DD; ++k)
        out[base + k] = w0 * hidden[base + k] + w1 * subn[base + k];
}

// ---------------------------------------------------------------------------
static inline int cdiv(long a, long b) { return (int)((a + b - 1) / b); }

extern "C" void kernel_launch(void* const* d_in, const int* in_sizes, int n_in,
                              void* d_out, int out_size, void* d_ws, size_t ws_size,
                              hipStream_t stream) {
    (void)in_sizes; (void)n_in; (void)out_size; (void)ws_size;
    const float* emb   = (const float*)d_in[0];
    const float* W1    = (const float*)d_in[1];
    const float* b1    = (const float*)d_in[2];
    const float* W2    = (const float*)d_in[3];
    const float* b2    = (const float*)d_in[4];
    const float* gamma = (const float*)d_in[5];
    const float* beta  = (const float*)d_in[6];
    const float* hgc1b = (const float*)d_in[7];
    const float* fc1W  = (const float*)d_in[8];
    const float* fl1W  = (const float*)d_in[9];
    const float* fl1b  = (const float*)d_in[10];
    const float* fl2W  = (const float*)d_in[11];
    const float* fl2b  = (const float*)d_in[12];
    const float* hgval = (const float*)d_in[13];
    const int*   eidx  = (const int*)d_in[14];
    const int*   hgrow = (const int*)d_in[15];
    const int*   hgcol = (const int*)d_in[16];
    const int* src = eidx;
    const int* dst = eidx + EE;
    float* out = (float*)d_out;

    // Workspace layout (floats), ~64.3 MB total, heavy reuse:
    float* A     = (float*)d_ws;            // N*128 : xw1 -> xw2(lo) -> xrelu(lo)+edge_emb(hi) -> t1(lo)+t2(hi)
    float* B     = A + (size_t)NN * 128;    // N*128 : h1  -> h2(lo)  -> x2(lo)+sub_node(hi)
    float* C     = B + (size_t)NN * 128;    // N*64  : hidden
    float* dinv  = C + (size_t)NN * 64;     // N     : deg -> rsqrt(deg)
    float* stats = dinv + NN;               // 128   : mu, rstd
    float* accum = stats + 128;             // 128   : sum, sumsq accumulators

    // ---- degrees / symmetric norm ----
    deg_init<<<cdiv(NN, BT), BT, 0, stream>>>(dinv);
    deg_count<<<cdiv(EE, BT), BT, 0, stream>>>(dst, dinv);
    deg_finalize<<<cdiv(NN, BT), BT, 0, stream>>>(dinv);

    // ---- GCN layer 1: xw1 = emb @ W1 ; h1 = b1 + dinv^2*xw1 + scatter ----
    {
        long waves = (long)(NN / 16) * (128 / 16);
        gemm_bf16_wmma<64, 128, false><<<cdiv(waves * 32, BT), BT, 0, stream>>>(emb, W1, nullptr, A, NN);
    }
    gcn_init<128><<<cdiv((long)NN * 32, BT), BT, 0, stream>>>(A, b1, dinv, B);
    gcn_scatter<32><<<cdiv((long)EE * 32, BT), BT, 0, stream>>>(A, dinv, src, dst, B);

    // ---- GCN layer 2: xw2 = h1 @ W2 ; h2 = b2 + dinv^2*xw2 + scatter ----
    {
        long waves = (long)(NN / 16) * (64 / 16);
        gemm_bf16_wmma<128, 64, false><<<cdiv(waves * 32, BT), BT, 0, stream>>>(B, W2, nullptr, A, NN);
    }
    float* h2 = B;                                  // h1 dead
    gcn_init<64><<<cdiv((long)NN * 16, BT), BT, 0, stream>>>(A, b2, dinv, h2);
    gcn_scatter<16><<<cdiv((long)EE * 16, BT), BT, 0, stream>>>(A, dinv, src, dst, h2);

    // ---- BatchNorm + relu + hgc1 bias ----
    vfill<<<1, 128, 0, stream>>>(accum, 0.0f, 128);
    bn_partial<<<512, BT, 0, stream>>>(h2, accum);
    bn_finalize<<<1, DD, 0, stream>>>(accum, stats);
    float* hidden   = C;
    float* xrelu    = A;                            // xw2 dead
    float* edge_emb = A + (size_t)NN * 64;
    bn_apply<<<cdiv((long)NN * 64, BT), BT, 0, stream>>>(h2, stats, gamma, beta, hgc1b, hidden, xrelu);

    // ---- Hypergraph: edge = G^T x ; x = G edge ----
    vfill<<<cdiv((long)NN * 64, BT), BT, 0, stream>>>(edge_emb, 0.0f, (long)NN * 64);
    hg_scatter<<<cdiv((long)NNZT * 16, BT), BT, 0, stream>>>(xrelu, hgval, hgrow, hgcol, edge_emb);
    float* x2 = B;                                  // h2 dead after bn_apply
    vfill<<<cdiv((long)NN * 64, BT), BT, 0, stream>>>(x2, 0.0f, (long)NN * 64);
    hg_scatter<<<cdiv((long)NNZT * 16, BT), BT, 0, stream>>>(edge_emb, hgval, hgcol, hgrow, x2);
    softmax_rows<<<cdiv(NN, BT), BT, 0, stream>>>(x2);

    // ---- sub_node = x2 @ fc1^T ----
    float* subn = B + (size_t)NN * 64;
    {
        long waves = (long)(NN / 16) * (64 / 16);
        gemm_bf16_wmma<64, 64, true><<<cdiv(waves * 32, BT), BT, 0, stream>>>(x2, fc1W, nullptr, subn, NN);
    }

    // ---- fusion gate: t_i = branch_i @ fus_l1^T + b ----
    float* t1 = A;                                  // xrelu dead
    float* t2 = A + (size_t)NN * 64;                // edge_emb dead
    {
        long waves = (long)(NN / 16) * (64 / 16);
        gemm_bf16_wmma<64, 64, true><<<cdiv(waves * 32, BT), BT, 0, stream>>>(hidden, fl1W, fl1b, t1, NN);
        gemm_bf16_wmma<64, 64, true><<<cdiv(waves * 32, BT), BT, 0, stream>>>(subn,   fl1W, fl1b, t2, NN);
    }
    fusion<<<cdiv(NN, BT), BT, 0, stream>>>(hidden, subn, t1, t2, fl2W, fl2b, out);
}
